// ResidualVectorQuantizer_16518444220717
// MI455X (gfx1250) — matrix operand order repaired and checked by reading, not compiled, not run on hardware
//
#include <hip/hip_runtime.h>
#include <stdint.h>

// ---------------- problem constants ----------------
#define N_ROWS  65536
#define DIM     128
#define STAGES  4
#define KCODE   2048
#define BETA_F  0.25f

#define ROWS_PER_BLOCK  128          // 8 waves x 16 rows
#define WAVES_PER_BLOCK 8
#define KCHUNK          64           // codes staged in LDS per block iteration
#define NCHUNKS         (KCODE / KCHUNK)            // 32 per stage
#define TCHUNKS         (STAGES * NCHUNKS)          // 128 total
#define RS              136          // padded bf16 residual row stride (elems), 272B = 17*16B
#define BS              72           // padded Bstage row stride (elems), 144B = 9*16B

typedef __bf16 bf16;
typedef __attribute__((ext_vector_type(16))) __bf16   v16bf;
typedef __attribute__((ext_vector_type(8)))  float    v8f;
typedef __attribute__((ext_vector_type(4)))  unsigned int u32x4;

union Frag16 { v16bf v; u32x4 q[2]; };   // 32B WMMA A/B operand

// async global->LDS copy, 16B per lane; same imm offset applies to LDS and global addr
#define ASYNC_B128(ldsv, vofs, sbase, OFF)                                     \
    asm volatile("global_load_async_to_lds_b128 %0, %1, %2 offset:" #OFF       \
                 :: "v"(ldsv), "v"(vofs), "s"(sbase) : "memory")

#define WAIT_ASYNCCNT(N)                                                       \
    asm volatile("s_wait_asynccnt " #N ::: "memory")

// ---------------- codebook norms: ||C_k||^2 in fp32 (argmin-exact term) ----------------
__global__ void cb_norms_kernel(const float* __restrict__ cb, float* __restrict__ norms) {
    const int wv   = (blockIdx.x * blockDim.x + threadIdx.x) >> 5;   // one wave per (s,k)
    const int lane = threadIdx.x & 31;
    const float* row = cb + (size_t)wv * DIM;
    const float4 v = *(const float4*)(row + lane * 4);
    float s = v.x * v.x + v.y * v.y + v.z * v.z + v.w * v.w;
    #pragma unroll
    for (int m = 16; m >= 1; m >>= 1) s += __shfl_xor(s, m, 32);
    if (lane == 0) norms[wv] = s;
}

// ---------------- codebook fp32 [S][K][D] -> bf16 transposed [S][D][K] ----------------
__global__ void cb_transpose_kernel(const float* __restrict__ cb, bf16* __restrict__ cbT) {
    __shared__ bf16 tile[32][33];
    const int s  = blockIdx.z;
    const int k0 = blockIdx.x * 32;
    const int d0 = blockIdx.y * 32;
    const int tx = threadIdx.x, ty = threadIdx.y;   // 32 x 8
    #pragma unroll
    for (int i = 0; i < 4; ++i) {
        const int ky = ty + 8 * i;
        tile[ky][tx] = (bf16)cb[((size_t)s * KCODE + (k0 + ky)) * DIM + d0 + tx];
    }
    __syncthreads();
    #pragma unroll
    for (int i = 0; i < 4; ++i) {
        const int dy = ty + 8 * i;
        cbT[((size_t)s * DIM + (d0 + dy)) * KCODE + k0 + tx] = tile[tx][dy];
    }
}

// ---------------- main RVQ kernel ----------------
__global__ __launch_bounds__(256)
void rvq_main_kernel(const float* __restrict__ x,
                     const float* __restrict__ cb,     // fp32 [S][K][D] for exact gather
                     const bf16*  __restrict__ cbT,    // bf16 [S][D][K]
                     const float* __restrict__ norms,  // fp32 [S][K]
                     float* __restrict__ out_xq,
                     int*   __restrict__ out_idx,
                     float* __restrict__ partials) {
    __shared__ bf16  resB[ROWS_PER_BLOCK * RS];       // bf16 residual mirror (WMMA A source)
    __shared__ bf16  Bstage[2][DIM * BS];             // double-buffered codebook chunk [d][k]
    __shared__ float normsStage[2][KCHUNK];
    __shared__ int   rowSel[WAVES_PER_BLOCK][16];

    const int tid      = threadIdx.x;
    const int lane     = tid & 31;
    const int wave     = tid >> 5;
    const int hlf      = lane >> 4;      // which 16-lane half
    const int mrow     = lane & 15;
    const int rowBase  = wave * 16;      // local row base for this wave
    const int gRowBase = blockIdx.x * ROWS_PER_BLOCK + rowBase;

    const uint64_t cbT_base = (uint64_t)(uintptr_t)cbT;
    const int dRow = tid >> 1;           // staging: 2 threads per d-row
    const int h2   = tid & 1;

    // fp32 residual lives in registers: resR[r] = residual[row r][lane*4 .. lane*4+3]
    float4 resR[16];
    #pragma unroll
    for (int r = 0; r < 16; ++r) {
        const float4 xv = *(const float4*)(x + (size_t)(gRowBase + r) * DIM + lane * 4);
        resR[r] = xv;
        bf16* bp = resB + (rowBase + r) * RS + lane * 4;
        bp[0] = (bf16)xv.x; bp[1] = (bf16)xv.y; bp[2] = (bf16)xv.z; bp[3] = (bf16)xv.w;
    }

    // ---- prologue: async-stage chunk t=0 into buffer 0 ----
    {
        const uint32_t vofs = (uint32_t)((dRow * KCODE + h2 * 32) * 2);
        const uint32_t ldsv = (uint32_t)(uintptr_t)(&Bstage[0][dRow * BS + h2 * 32]);
        ASYNC_B128(ldsv, vofs, cbT_base, 0);
        ASYNC_B128(ldsv, vofs, cbT_base, 16);
        ASYNC_B128(ldsv, vofs, cbT_base, 32);
        ASYNC_B128(ldsv, vofs, cbT_base, 48);
        if (tid < KCHUNK) normsStage[0][tid] = norms[tid];
    }

    for (int s = 0; s < STAGES; ++s) {
        // A fragments (16x32 bf16, documented gfx1250 lane layout), register-resident per stage
        Frag16 a[4];
        {
            const bf16* rp = resB + (rowBase + mrow) * RS;
            #pragma unroll
            for (int c = 0; c < 4; ++c) {
                const int db = c * 32;
                a[c].q[0] = *(const u32x4*)(rp + db + hlf * 8);
                a[c].q[1] = *(const u32x4*)(rp + db + 16 + hlf * 8);
            }
        }
        float runval[8];
        int   runidx[8];
        #pragma unroll
        for (int j = 0; j < 8; ++j) { runval[j] = 3.0e38f; runidx[j] = 0; }

        for (int kbi = 0; kbi < NCHUNKS; ++kbi) {
            const int t   = s * NCHUNKS + kbi;     // flattened chunk id
            const int cur = t & 1;
            const int nxt = cur ^ 1;
            const int kb  = kbi * KCHUNK;

            // all waves done computing on buf[nxt] (used at t-1) before refilling it
            __syncthreads();
            if (t + 1 < TCHUNKS) {
                // next chunk may cross a stage boundary: global element
                //   ((s_next*DIM + d) * KCODE + kb_next)  ==  flattened via t+1
                const int sn  = (t + 1) >> 5;              // NCHUNKS == 32
                const int kbn = ((t + 1) & 31) * KCHUNK;
                const uint32_t vofs =
                    (uint32_t)((((sn * DIM + dRow) * KCODE) + kbn + h2 * 32) * 2);
                const uint32_t ldsv =
                    (uint32_t)(uintptr_t)(&Bstage[nxt][dRow * BS + h2 * 32]);
                ASYNC_B128(ldsv, vofs, cbT_base, 0);
                ASYNC_B128(ldsv, vofs, cbT_base, 16);
                ASYNC_B128(ldsv, vofs, cbT_base, 32);
                ASYNC_B128(ldsv, vofs, cbT_base, 48);
                if (tid < KCHUNK) normsStage[nxt][tid] = norms[(t + 1) * KCHUNK + tid];
                // in-order async completion: <=4 outstanding => chunk t's 4 ops retired
                WAIT_ASYNCCNT(4);
            } else {
                WAIT_ASYNCCNT(0);
            }
            __syncthreads();                       // publish buf[cur] to all waves

            #pragma unroll
            for (int kc = 0; kc < KCHUNK; kc += 16) {
                v8f acc = {0.f, 0.f, 0.f, 0.f, 0.f, 0.f, 0.f, 0.f};
                #pragma unroll
                for (int c = 0; c < 4; ++c) {      // accumulate over D=128 in 32-chunks
                    Frag16 b;
                    const bf16* bp = &Bstage[cur][(c * 32 + hlf * 8) * BS + kc + mrow];
                    #pragma unroll
                    for (int i = 0; i < 8; ++i) {
                        b.v[i]     = bp[i * BS];
                        b.v[8 + i] = bp[(16 + i) * BS];
                    }
                    acc = __builtin_amdgcn_wmma_f32_16x16x32_bf16(
                              false, a[c].v, false, b.v, (short)0, acc, false, false);
                }
                const float nrm = normsStage[cur][kc + mrow];
                const int   kab = kb + kc + mrow;
                #pragma unroll
                for (int j = 0; j < 8; ++j) {      // acc[j] = row (j or j+8), code = lane&15
                    const float sc = __builtin_fmaf(-2.0f, acc[j], nrm);
                    if (sc < runval[j]) { runval[j] = sc; runidx[j] = kab; }
                }
            }
        }

        // argmin across 16 lanes in each half (masks <16 stay within a half)
        #pragma unroll
        for (int j = 0; j < 8; ++j) {
            #pragma unroll
            for (int m = 8; m >= 1; m >>= 1) {
                const float ov = __shfl_xor(runval[j], m, 32);
                const int   oi = __shfl_xor(runidx[j], m, 32);
                if (ov < runval[j] || (ov == runval[j] && oi < runidx[j])) {
                    runval[j] = ov; runidx[j] = oi;
                }
            }
        }
        if (lane == 0 || lane == 16) {             // lane0 -> rows 0..7, lane16 -> rows 8..15
            const int rb = hlf * 8;
            #pragma unroll
            for (int j = 0; j < 8; ++j) {
                rowSel[wave][rb + j] = runidx[j];
                out_idx[(size_t)(gRowBase + rb + j) * STAGES + s] = runidx[j];
            }
        }

        // exact fp32 gather + residual update; loss term = ||new residual||^2
        const float* cb_s = cb + (size_t)s * KCODE * DIM;
        float lossAcc = 0.0f;
        #pragma unroll
        for (int r = 0; r < 16; ++r) {
            const int idx = rowSel[wave][r];
            const float4 c4 = *(const float4*)(cb_s + (size_t)idx * DIM + lane * 4);
            float4 n4;
            n4.x = resR[r].x - c4.x; n4.y = resR[r].y - c4.y;
            n4.z = resR[r].z - c4.z; n4.w = resR[r].w - c4.w;
            resR[r] = n4;
            bf16* bp = resB + (rowBase + r) * RS + lane * 4;
            bp[0] = (bf16)n4.x; bp[1] = (bf16)n4.y; bp[2] = (bf16)n4.z; bp[3] = (bf16)n4.w;
            lossAcc += n4.x * n4.x + n4.y * n4.y + n4.z * n4.z + n4.w * n4.w;
        }
        #pragma unroll
        for (int m = 16; m >= 1; m >>= 1) lossAcc += __shfl_xor(lossAcc, m, 32);
        if (lane == 0)
            partials[(size_t)s * (N_ROWS / 16) + blockIdx.x * WAVES_PER_BLOCK + wave] = lossAcc;
    }

    // x_q = x - final residual (sum of selected codewords)
    #pragma unroll
    for (int r = 0; r < 16; ++r) {
        const size_t go = (size_t)(gRowBase + r) * DIM + lane * 4;
        const float4 xv = *(const float4*)(x + go);
        float4 o;
        o.x = xv.x - resR[r].x; o.y = xv.y - resR[r].y;
        o.z = xv.z - resR[r].z; o.w = xv.w - resR[r].w;
        *(float4*)(out_xq + go) = o;
    }
}

// ---------------- deterministic loss reduction ----------------
__global__ void rvq_finalize_kernel(const float* __restrict__ partials,
                                    float* __restrict__ out_loss) {
    __shared__ float red[256];
    float s = 0.0f;
    const int total = STAGES * (N_ROWS / 16);
    for (int i = threadIdx.x; i < total; i += 256) s += partials[i];
    red[threadIdx.x] = s;
    __syncthreads();
    for (int m = 128; m >= 1; m >>= 1) {
        if ((int)threadIdx.x < m) red[threadIdx.x] += red[threadIdx.x + m];
        __syncthreads();
    }
    if (threadIdx.x == 0)
        out_loss[0] = red[0] * ((1.0f + BETA_F) /
                                ((float)STAGES * (float)N_ROWS * (float)DIM));
}

// ---------------- launcher ----------------
extern "C" void kernel_launch(void* const* d_in, const int* in_sizes, int n_in,
                              void* d_out, int out_size, void* d_ws, size_t ws_size,
                              hipStream_t stream) {
    const float* x  = (const float*)d_in[0];          // [N, D] fp32
    const float* cb = (const float*)d_in[1];          // [S, K, D] fp32

    char* ws = (char*)d_ws;
    bf16*  cbT      = (bf16*)ws;                                        // 2 MB
    float* norms    = (float*)(ws + (size_t)STAGES * DIM * KCODE * 2);  // 32 KB
    float* partials = (float*)(ws + (size_t)STAGES * DIM * KCODE * 2
                                  + (size_t)STAGES * KCODE * 4);        // 64 KB

    float* out_xq   = (float*)d_out;
    float* out_loss = out_xq + (size_t)N_ROWS * DIM;
    int*   out_idx  = (int*)(out_xq + (size_t)N_ROWS * DIM + 1);

    cb_norms_kernel<<<(STAGES * KCODE) / 8, 256, 0, stream>>>(cb, norms);
    cb_transpose_kernel<<<dim3(KCODE / 32, DIM / 32, STAGES), dim3(32, 8, 1), 0, stream>>>(cb, cbT);
    rvq_main_kernel<<<N_ROWS / ROWS_PER_BLOCK, 256, 0, stream>>>(
        x, cb, cbT, norms, out_xq, out_idx, partials);
    rvq_finalize_kernel<<<1, 256, 0, stream>>>(partials, out_loss);
}
